// SubMAttention3d_57561151701634
// MI455X (gfx1250) — matrix operand order repaired and checked by reading, not compiled, not run on hardware
//
#include <hip/hip_runtime.h>
#include <hip/hip_bf16.h>
#include <cstdint>

// ---------------------------------------------------------------------------
// Problem constants (from reference): N=50000, K=48, C=128, H=8, D=16, FF=256, OUT=128
// ---------------------------------------------------------------------------
#define KK   48
#define CC   128
#define HH   8
#define DD   16
#define FFN  256

typedef __attribute__((ext_vector_type(16))) _Float16 v16h;
typedef __attribute__((ext_vector_type(8)))  _Float16 v8h;
typedef __attribute__((ext_vector_type(8)))  float    v8f;

// Padded LDS strides (in halves) to rotate banks between rows.
#define KSTR 136   // 48 x 128 key-feature / k / v tiles
#define XSTR 136   // 16 x 128 activation tiles
#define HSTR 264   // 16 x 256 hidden tile

// f16 weight arena layout inside d_ws (element offsets, _Float16 units):
//  [0      .. 16384) wq      (128x128)
//  [16384  .. 32768) wk      (128x128)
//  [32768  .. 49152) wv      (128x128)
//  [49152  .. 65536) out_proj(128x128)
//  [65536  .. 98304) ff1     (256x128)
//  [98304  ..131072) ff2     (128x256)
//  [131072 ..147456) out_w   (128x128)
#define W_WQ   0
#define W_WK   16384
#define W_WV   32768
#define W_WO   49152
#define W_FF1  65536
#define W_FF2  98304
#define W_OUT  131072
#define W_TOT  147456

// Build a 16x32 f16 A-fragment from two contiguous 8-half LDS segments
// (K = base..base+7 and base+8..base+15, contiguous k order).
static __device__ inline v16h ld_a_frag(const _Float16* p) {
  v8h lo = *(const v8h*)p;
  v8h hi = *(const v8h*)(p + 8);
  v16h r;
#pragma unroll
  for (int i = 0; i < 8; ++i) { r[i] = lo[i]; r[i + 8] = hi[i]; }
  return r;
}

static __device__ inline float wave_sum(float v) {
#pragma unroll
  for (int off = 16; off > 0; off >>= 1) v += __shfl_xor(v, off, 32);
  return v;
}
static __device__ inline float wave_max(float v) {
#pragma unroll
  for (int off = 16; off > 0; off >>= 1) v = fmaxf(v, __shfl_xor(v, off, 32));
  return v;
}

// ---------------------------------------------------------------------------
// Kernel 0: one-shot f32 -> f16 weight conversion into the d_ws arena.
// ---------------------------------------------------------------------------
__global__ void convert_weights(const float* __restrict__ in_proj_w,
                                const float* __restrict__ out_proj_w,
                                const float* __restrict__ ff1_w,
                                const float* __restrict__ ff2_w,
                                const float* __restrict__ out_w,
                                _Float16* __restrict__ dst) {
  int i = blockIdx.x * 256 + threadIdx.x;
  if (i >= W_TOT) return;
  float v;
  if      (i < W_WO)   v = in_proj_w[i];            // wq|wk|wv contiguous
  else if (i < W_FF1)  v = out_proj_w[i - W_WO];
  else if (i < W_FF2)  v = ff1_w[i - W_FF1];
  else if (i < W_OUT)  v = ff2_w[i - W_FF2];
  else                 v = out_w[i - W_OUT];
  dst[i] = (_Float16)v;
}

// ---------------------------------------------------------------------------
// Kernel 1: attention. One workgroup (256 thr / 8 waves) per voxel.
//   gather+posemb -> LDS f16, K/V projection via WMMA (192 wmma/block),
//   q VALU, scores, wave-per-head softmax, ctx, out_proj, LN1.
//   Writes x (N x 128 f32) into xout (staged in d_out).
// ---------------------------------------------------------------------------
__global__ __launch_bounds__(256) void attn_kernel(
    const float* __restrict__ vf, const float* __restrict__ coords,
    const int* __restrict__ kidx, const uint8_t* __restrict__ kmask,
    const float* __restrict__ in_proj_b, const float* __restrict__ out_proj_b,
    const float* __restrict__ kpw, const float* __restrict__ kpb,
    const float* __restrict__ ln1g, const float* __restrict__ ln1b,
    const _Float16* __restrict__ w16, float* __restrict__ xout) {
  __shared__ _Float16 kf[KK * KSTR];    // gathered key feats (+pos emb), f16
  __shared__ _Float16 kk16[KK * KSTR];  // K projection result, f16
  __shared__ _Float16 vv16[KK * KSTR];  // V projection result, f16
  __shared__ float rel[KK][3];
  __shared__ int   idxs[KK];
  __shared__ float vfn[CC];
  __shared__ float qf[CC];
  __shared__ float sc[HH * KK];
  __shared__ float ctxs[CC];
  __shared__ float ybuf[CC];
  __shared__ float red[2];

  const int n    = blockIdx.x;
  const int tid  = threadIdx.x;
  const int lane = tid & 31;
  const int wv   = tid >> 5;           // wave id 0..7
  const int lrow = lane & 15;
  const int half = (lane >> 4) & 1;

  // ---- phase 0: indices, relative coords, residual copy -------------------
  if (tid < KK) {
    int id = kidx[n * KK + tid];
    idxs[tid] = id;
    float cx = coords[n * 3 + 0], cy = coords[n * 3 + 1], cz = coords[n * 3 + 2];
    rel[tid][0] = coords[id * 3 + 0] - cx;
    rel[tid][1] = coords[id * 3 + 1] - cy;
    rel[tid][2] = coords[id * 3 + 2] - cz;
  }
  if (tid < CC) vfn[tid] = vf[(size_t)n * CC + tid];
  __syncthreads();

  // ---- phase 1: gather key features + relu(pos emb), to f16 LDS -----------
  for (int i = tid; i < KK * CC; i += 256) {
    int k = i >> 7, c = i & 127;
    int id = idxs[k];
    float f  = vf[(size_t)id * CC + c];
    float pe = rel[k][0] * kpw[c * 3 + 0] + rel[k][1] * kpw[c * 3 + 1] +
               rel[k][2] * kpw[c * 3 + 2] + kpb[c];
    pe = fmaxf(pe, 0.0f);
    kf[k * KSTR + c] = (_Float16)(f + pe);
  }
  __syncthreads();

  // ---- phase 2: K and V projections with WMMA -----------------------------
  // out(48x128) = kf(48x128) @ W^T ; tiles: m in 0..2, nn in 0..7, 4 k-steps.
#pragma unroll
  for (int p = 0; p < 2; ++p) {
    const _Float16* W    = w16 + (p ? W_WV : W_WK);
    const float*    bias = in_proj_b + (p ? 2 * CC : CC);
    _Float16*       dst  = p ? vv16 : kk16;
    for (int t = wv; t < 24; t += 8) {
      int m = t >> 3, nn = t & 7;
      float b = bias[nn * 16 + lrow];
      v8f acc = {b, b, b, b, b, b, b, b};
#pragma unroll
      for (int ks = 0; ks < 4; ++ks) {
        int k0 = ks * 32;
        v16h a  = ld_a_frag(&kf[(m * 16 + lrow) * KSTR + k0 + half * 16]);
        v16h bf = *(const v16h*)(W + (nn * 16 + lrow) * CC + k0 + half * 16);
        acc = __builtin_amdgcn_wmma_f32_16x16x32_f16(false, a, false, bf,
                                                     (short)0, acc, false, false);
      }
#pragma unroll
      for (int i = 0; i < 8; ++i) {
        int row = m * 16 + i + half * 8;
        dst[row * KSTR + nn * 16 + lrow] = (_Float16)acc[i];
      }
    }
  }

  // ---- phase 3: q projection (single row, VALU), scaled by 1/sqrt(D) ------
  if (tid < CC) {
    const _Float16* wq = w16 + W_WQ + tid * CC;
    float s = in_proj_b[tid];
#pragma unroll 8
    for (int c = 0; c < CC; ++c) s += vfn[c] * (float)wq[c];
    qf[tid] = s * 0.25f;  // 1/sqrt(16)
  }
  __syncthreads();

  // ---- phase 4: scores (h,k) dots over D=16, apply mask -------------------
  for (int i = tid; i < HH * KK; i += 256) {
    int h = i / KK, k = i - h * KK;
    float s = 0.0f;
#pragma unroll
    for (int d = 0; d < DD; ++d)
      s += qf[h * DD + d] * (float)kk16[k * KSTR + h * DD + d];
    if (kmask[n * KK + k]) s = -1.0e9f;
    sc[h * KK + k] = s;
  }
  __syncthreads();

  // ---- phase 5: softmax, one wave per head (48 keys -> 1.5 per lane) ------
  {
    int h = wv;
    float s0 = sc[h * KK + lane];
    float s1 = (lane < 16) ? sc[h * KK + 32 + lane] : -3.0e38f;
    float m  = wave_max(fmaxf(s0, s1));
    float e0 = __expf(s0 - m);
    float e1 = (lane < 16) ? __expf(s1 - m) : 0.0f;
    float inv = 1.0f / wave_sum(e0 + e1);
    sc[h * KK + lane] = e0 * inv;
    if (lane < 16) sc[h * KK + 32 + lane] = e1 * inv;
  }
  __syncthreads();

  // ---- phase 6: ctx = attn @ v --------------------------------------------
  if (tid < CC) {
    int h = tid >> 4;
    float s = 0.0f;
#pragma unroll 8
    for (int k = 0; k < KK; ++k)
      s += sc[h * KK + k] * (float)vv16[k * KSTR + tid];
    ctxs[tid] = s;
  }
  __syncthreads();

  // ---- phase 7: out_proj + residual ---------------------------------------
  if (tid < CC) {
    const _Float16* wo = w16 + W_WO + tid * CC;
    float s = out_proj_b[tid];
#pragma unroll 8
    for (int c = 0; c < CC; ++c) s += ctxs[c] * (float)wo[c];
    ybuf[tid] = vfn[tid] + s;
  }
  __syncthreads();

  // ---- phase 8: LayerNorm-1 stats (wave 0) then normalize + store x -------
  if (wv == 0) {
    float s = 0.0f, q2 = 0.0f;
#pragma unroll
    for (int j = 0; j < 4; ++j) {
      float y = ybuf[lane + 32 * j];
      s += y; q2 += y * y;
    }
    s = wave_sum(s); q2 = wave_sum(q2);
    if (lane == 0) {
      float mean = s * (1.0f / CC);
      float var  = q2 * (1.0f / CC) - mean * mean;
      red[0] = mean;
      red[1] = rsqrtf(var + 1e-5f);
    }
  }
  __syncthreads();
  if (tid < CC) {
    float o = (ybuf[tid] - red[0]) * red[1] * ln1g[tid] + ln1b[tid];
    xout[(size_t)n * CC + tid] = o;
  }
}

// ---------------------------------------------------------------------------
// Kernel 2: FFN + final projection. 16 voxels per workgroup, all-WMMA GEMMs.
//   x -> relu(x@ff1^T+b) -> @ff2^T+b + x -> LN2 -> @out_w^T+b -> LN3 -> relu
//   Reads/writes the same 16 rows of xio (in-place safe, barrier ordered).
// ---------------------------------------------------------------------------
__global__ __launch_bounds__(256) void ffn_kernel(
    const float* __restrict__ ff1b, const float* __restrict__ ff2b,
    const float* __restrict__ outb,
    const float* __restrict__ ln2g, const float* __restrict__ ln2b,
    const float* __restrict__ ln3g, const float* __restrict__ ln3b,
    const _Float16* __restrict__ w16, float* __restrict__ xio, int Ntot) {
  __shared__ _Float16 xh[16 * XSTR];
  __shared__ float    xf[16 * CC];
  __shared__ _Float16 hh[16 * HSTR];
  __shared__ float    yf[16 * CC];
  __shared__ _Float16 x2h[16 * XSTR];
  __shared__ float    zf[16 * CC];

  const int tid  = threadIdx.x;
  const int lane = tid & 31;
  const int wv   = tid >> 5;
  const int lrow = lane & 15;
  const int half = (lane >> 4) & 1;
  const long base = (long)blockIdx.x * 16;

  // ---- load x tile --------------------------------------------------------
  for (int i = tid; i < 16 * CC; i += 256) {
    int r = i >> 7, c = i & 127;
    long g = base + r;
    float v = (g < Ntot) ? xio[g * CC + c] : 0.0f;
    xf[r * CC + c]   = v;
    xh[r * XSTR + c] = (_Float16)v;
  }
  __syncthreads();

  // ---- GEMM1: h = relu(x @ ff1^T + b)  (16x128)@(128x256) -----------------
  {
    const _Float16* W1 = w16 + W_FF1;
    for (int t = wv; t < 16; t += 8) {
      float b = ff1b[t * 16 + lrow];
      v8f acc = {b, b, b, b, b, b, b, b};
#pragma unroll
      for (int ks = 0; ks < 4; ++ks) {
        int k0 = ks * 32;
        v16h a  = ld_a_frag(&xh[lrow * XSTR + k0 + half * 16]);
        v16h bf = *(const v16h*)(W1 + (t * 16 + lrow) * CC + k0 + half * 16);
        acc = __builtin_amdgcn_wmma_f32_16x16x32_f16(false, a, false, bf,
                                                     (short)0, acc, false, false);
      }
#pragma unroll
      for (int i = 0; i < 8; ++i) {
        int row = i + half * 8;
        hh[row * HSTR + t * 16 + lrow] = (_Float16)fmaxf(acc[i], 0.0f);
      }
    }
  }
  __syncthreads();

  // ---- GEMM2: y = h @ ff2^T + b + x  (16x256)@(256x128) -------------------
  {
    const _Float16* W2 = w16 + W_FF2;
    int t = wv;  // 8 tiles, one per wave
    float b = ff2b[t * 16 + lrow];
    v8f acc = {b, b, b, b, b, b, b, b};
#pragma unroll
    for (int ks = 0; ks < 8; ++ks) {
      int k0 = ks * 32;
      v16h a  = ld_a_frag(&hh[lrow * HSTR + k0 + half * 16]);
      v16h bf = *(const v16h*)(W2 + (t * 16 + lrow) * FFN + k0 + half * 16);
      acc = __builtin_amdgcn_wmma_f32_16x16x32_f16(false, a, false, bf,
                                                   (short)0, acc, false, false);
    }
#pragma unroll
    for (int i = 0; i < 8; ++i) {
      int row = i + half * 8, col = t * 16 + lrow;
      yf[row * CC + col] = acc[i] + xf[row * CC + col];
    }
  }
  __syncthreads();

  // ---- LN2 per row (wave handles 2 rows), emit f16 x2 ---------------------
#pragma unroll
  for (int rr = 0; rr < 2; ++rr) {
    int r = wv * 2 + rr;
    float s = 0.0f, q2 = 0.0f;
#pragma unroll
    for (int j = 0; j < 4; ++j) {
      float y = yf[r * CC + lane + 32 * j];
      s += y; q2 += y * y;
    }
    s = wave_sum(s); q2 = wave_sum(q2);
    float mean = s * (1.0f / CC);
    float rstd = rsqrtf(q2 * (1.0f / CC) - mean * mean + 1e-5f);
#pragma unroll
    for (int j = 0; j < 4; ++j) {
      int c = lane + 32 * j;
      float v = (yf[r * CC + c] - mean) * rstd * ln2g[c] + ln2b[c];
      x2h[r * XSTR + c] = (_Float16)v;
    }
  }
  __syncthreads();

  // ---- GEMM3: z = x2 @ out_w^T + b  (16x128)@(128x128) --------------------
  {
    const _Float16* W3 = w16 + W_OUT;
    int t = wv;
    float b = outb[t * 16 + lrow];
    v8f acc = {b, b, b, b, b, b, b, b};
#pragma unroll
    for (int ks = 0; ks < 4; ++ks) {
      int k0 = ks * 32;
      v16h a  = ld_a_frag(&x2h[lrow * XSTR + k0 + half * 16]);
      v16h bf = *(const v16h*)(W3 + (t * 16 + lrow) * CC + k0 + half * 16);
      acc = __builtin_amdgcn_wmma_f32_16x16x32_f16(false, a, false, bf,
                                                   (short)0, acc, false, false);
    }
#pragma unroll
    for (int i = 0; i < 8; ++i)
      zf[(i + half * 8) * CC + t * 16 + lrow] = acc[i];
  }
  __syncthreads();

  // ---- LN3 + relu + writeback --------------------------------------------
#pragma unroll
  for (int rr = 0; rr < 2; ++rr) {
    int r = wv * 2 + rr;
    long g = base + r;
    float s = 0.0f, q2 = 0.0f;
#pragma unroll
    for (int j = 0; j < 4; ++j) {
      float z = zf[r * CC + lane + 32 * j];
      s += z; q2 += z * z;
    }
    s = wave_sum(s); q2 = wave_sum(q2);
    float mean = s * (1.0f / CC);
    float rstd = rsqrtf(q2 * (1.0f / CC) - mean * mean + 1e-5f);
    if (g < Ntot) {
#pragma unroll
      for (int j = 0; j < 4; ++j) {
        int c = lane + 32 * j;
        float o = (zf[r * CC + c] - mean) * rstd * ln3g[c] + ln3b[c];
        xio[g * CC + c] = fmaxf(o, 0.0f);
      }
    }
  }
}

// ---------------------------------------------------------------------------
extern "C" void kernel_launch(void* const* d_in, const int* in_sizes, int n_in,
                              void* d_out, int out_size, void* d_ws, size_t ws_size,
                              hipStream_t stream) {
  const float*   vf         = (const float*)d_in[0];
  const float*   coords     = (const float*)d_in[1];
  const int*     kidx       = (const int*)d_in[2];
  const uint8_t* kmask      = (const uint8_t*)d_in[3];
  const float*   in_proj_w  = (const float*)d_in[4];
  const float*   in_proj_b  = (const float*)d_in[5];
  const float*   out_proj_w = (const float*)d_in[6];
  const float*   out_proj_b = (const float*)d_in[7];
  const float*   kpw        = (const float*)d_in[8];
  const float*   kpb        = (const float*)d_in[9];
  const float*   ln1g       = (const float*)d_in[10];
  const float*   ln1b       = (const float*)d_in[11];
  const float*   ln2g       = (const float*)d_in[12];
  const float*   ln2b       = (const float*)d_in[13];
  const float*   ff1w       = (const float*)d_in[14];
  const float*   ff1b       = (const float*)d_in[15];
  const float*   ff2w       = (const float*)d_in[16];
  const float*   ff2b       = (const float*)d_in[17];
  const float*   outw       = (const float*)d_in[18];
  const float*   outb       = (const float*)d_in[19];
  const float*   ln3g       = (const float*)d_in[20];
  const float*   ln3b       = (const float*)d_in[21];

  float*     out = (float*)d_out;
  _Float16*  w16 = (_Float16*)d_ws;
  const int  N   = in_sizes[0] / CC;

  convert_weights<<<(W_TOT + 255) / 256, 256, 0, stream>>>(
      in_proj_w, out_proj_w, ff1w, ff2w, outw, w16);

  attn_kernel<<<N, 256, 0, stream>>>(vf, coords, kidx, kmask, in_proj_b,
                                     out_proj_b, kpw, kpb, ln1g, ln1b, w16, out);

  ffn_kernel<<<(N + 15) / 16, 256, 0, stream>>>(ff1b, ff2b, outb, ln2g, ln2b,
                                                ln3g, ln3b, w16, out, N);
}
